// BiGRU_VERT_HORIZ_84215718740513
// MI455X (gfx1250) — compile-verified, matrix operand order checked by reading, not run on hardware
//
#include <hip/hip_runtime.h>
#include <hip/hip_bf16.h>
#include <math.h>

// Problem dims (compile-time constants from the reference)
#define BB 32    // batch
#define SS 128   // sequence (inner scan)
#define CC 128   // columns (outer scan)
#define HH 128   // hidden
#define OO 64    // output classes

typedef __attribute__((ext_vector_type(16))) _Float16 v16h;
typedef __attribute__((ext_vector_type(8)))  float    v8f;

__device__ __forceinline__ float fast_sigmoid(float x) {
    return 1.0f / (1.0f + __expf(-x));
}
__device__ __forceinline__ float fast_tanh(float x) {
    x = fminf(fmaxf(x, -15.0f), 15.0f);
    float e = __expf(2.0f * x);
    return (e - 1.0f) / (e + 1.0f);
}

union AFrag { v16h v; float4 f2[2]; };

// One block per direction. 256 threads = 8 waves (wave32).
// Wave w owns hidden columns j in [16w, 16w+16) for ALL THREE gates, so the
// gate nonlinearity + h update are wave-local (no cross-wave gh staging).
// Per step: 24 x v_wmma_f32_16x16x32_f16 per wave, one s_barrier (ping-pong h).
__global__ void __launch_bounds__(256)
gru_chain_kernel(const float* __restrict__ x,
                 const float* __restrict__ h_prev,
                 const float* __restrict__ Wih_f, const float* __restrict__ Whh_f,
                 const float* __restrict__ bih_f, const float* __restrict__ bhh_f,
                 const float* __restrict__ Wih_b, const float* __restrict__ Whh_b,
                 const float* __restrict__ bih_b, const float* __restrict__ bhh_b,
                 float* __restrict__ feat)
{
    const int dir  = blockIdx.x;          // 0 = forward, 1 = backward
    const int tid  = threadIdx.x;
    const int lane = tid & 31;
    const int w    = tid >> 5;            // wave id 0..7
    const int ln   = lane & 15;
    const int hi   = lane >> 4;           // half-wave select (K/M +8 offset)
    const int jl   = 16 * w + ln;         // this lane's hidden column

    const float* Wih = dir ? Wih_b : Wih_f;
    const float* Whh = dir ? Whh_b : Whh_f;
    const float* bih = dir ? bih_b : bih_f;
    const float* bhh = dir ? bhh_b : bhh_f;

    __shared__ __align__(16) _Float16 h16[2][BB * HH];  // 16 KB (WMMA A source)
    __shared__ float                  h32[2][BB * HH];  // 32 KB (exact h for z*h)
    __shared__ float                  xs[BB * SS];      // 16 KB (staged x column)

    // ---- Pre-pack B fragments (Whh^T) into registers as f16 ----------------
    // gh[b][n] = sum_k h[b][k] * Whh[n][k]  =>  B(k,n) = Whh[n][k].
    // B-fragment layout mirrors the ISA 16-bit A layout: lane holds N = lane&15,
    // VGPR v holds K pair {k,k+1} with k = 32*kt + 8*(lane>=16) + 2v (v<4)
    // or 16 + ... (v>=4).
    v16h Bf[3][4];
#pragma unroll
    for (int g = 0; g < 3; ++g) {
        const float* wrow = Whh + (g * HH + jl) * HH;
#pragma unroll
        for (int kt = 0; kt < 4; ++kt) {
            v16h f;
#pragma unroll
            for (int v = 0; v < 8; ++v) {
                int k = 32 * kt + 8 * hi + ((v < 4) ? 2 * v : 16 + 2 * (v - 4));
                f[2 * v]     = (_Float16)wrow[k];
                f[2 * v + 1] = (_Float16)wrow[k + 1];
            }
            Bf[g][kt] = f;
        }
    }

    // Per-lane gate constants (Wih is (3H,1) -> scalar per gate per column)
    float wih_g[3], bih_g[3], bhh_g[3];
#pragma unroll
    for (int g = 0; g < 3; ++g) {
        wih_g[g] = Wih[g * HH + jl];
        bih_g[g] = bih[g * HH + jl];
        bhh_g[g] = bhh[g * HH + jl];
    }

    // Init hidden state ping-pong buffer 0 from h_prev
    for (int i = tid; i < BB * HH; i += 256) {
        float v = h_prev[dir * BB * HH + i];
        h32[0][i] = v;
        h16[0][i] = (_Float16)v;
    }

    int p = 0;
    for (int c = 0; c < CC; ++c) {
        // Stage x[:, :, c] into LDS (reused by 128 steps)
        for (int i = tid; i < BB * SS; i += 256) {
            int b = i >> 7, s = i & (SS - 1);
            xs[i] = x[(b * SS + s) * CC + c];
        }
        __syncthreads();

        for (int s = 0; s < SS; ++s) {
            const int st = dir ? (SS - 1 - s) : s;   // backward reads reversed seq

            // ---- gh = h @ Whh^T for this wave's 16 columns, 3 gates --------
            v8f acc[3][2];
            v8f zero = {};
#pragma unroll
            for (int g = 0; g < 3; ++g) { acc[g][0] = zero; acc[g][1] = zero; }

#pragma unroll
            for (int kt = 0; kt < 4; ++kt) {
                // A fragments per ISA 16-bit 16x32 layout: lane M = lane&15,
                // K = 32*kt + 8*hi + {0..7} (v0..3) and +16 (v4..7): 2x b128.
                AFrag a0, a1;
                const _Float16* r0 = &h16[p][ ln       * HH + 32 * kt + 8 * hi];
                const _Float16* r1 = &h16[p][(16 + ln) * HH + 32 * kt + 8 * hi];
                a0.f2[0] = *(const float4*)r0;
                a0.f2[1] = *(const float4*)(r0 + 16);
                a1.f2[0] = *(const float4*)r1;
                a1.f2[1] = *(const float4*)(r1 + 16);
#pragma unroll
                for (int g = 0; g < 3; ++g) {
                    acc[g][0] = __builtin_amdgcn_wmma_f32_16x16x32_f16(
                        false, a0.v, false, Bf[g][kt], (short)0, acc[g][0], false, false);
                    acc[g][1] = __builtin_amdgcn_wmma_f32_16x16x32_f16(
                        false, a1.v, false, Bf[g][kt], (short)0, acc[g][1], false, false);
                }
            }

            // ---- gates + h update; write ping-pong partner buffer ----------
            // D-tile layout: VGPR v, lane -> row b = 16m + v + 8*hi, col jl.
#pragma unroll
            for (int m = 0; m < 2; ++m) {
#pragma unroll
                for (int v = 0; v < 8; ++v) {
                    int b      = 16 * m + 8 * hi + v;
                    float xt   = xs[b * SS + st];
                    float hold = h32[p][b * HH + jl];
                    float hr = acc[0][m][v] + bhh_g[0];
                    float hz = acc[1][m][v] + bhh_g[1];
                    float hn = acc[2][m][v] + bhh_g[2];
                    float ir = fmaf(xt, wih_g[0], bih_g[0]);
                    float iz = fmaf(xt, wih_g[1], bih_g[1]);
                    float in_= fmaf(xt, wih_g[2], bih_g[2]);
                    float r  = fast_sigmoid(ir + hr);
                    float z  = fast_sigmoid(iz + hz);
                    float n  = fast_tanh(fmaf(r, hn, in_));
                    float hnew = fmaf(z, hold - n, n);   // (1-z)*n + z*h
                    h32[p ^ 1][b * HH + jl] = hnew;
                    h16[p ^ 1][b * HH + jl] = (_Float16)hnew;
                }
            }
            __syncthreads();   // single barrier per step (ping-pong buffers)
            p ^= 1;
        }

        // End-of-column features: feat[c][b][dir*H + j] = h[b][j]
        for (int i = tid; i < BB * HH; i += 256) {
            int b = i >> 7, j = i & (HH - 1);
            feat[(c * BB + b) * (2 * HH) + dir * HH + j] = h32[p][i];
        }
        // next loop's post-x-stage __syncthreads orders this copy vs. updates
    }
}

// FC (256 -> 64) + relu + softmax; one block per column c. 134 MFLOP total.
__global__ void __launch_bounds__(256)
fc_softmax_kernel(const float* __restrict__ feat,
                  const float* __restrict__ Wfc,
                  const float* __restrict__ bfc,
                  float* __restrict__ out)
{
    const int c = blockIdx.x;
    const int t = threadIdx.x;
    __shared__ float sf[BB * 2 * HH];   // 32 KB
    __shared__ float logits[BB][OO];    // 8 KB

    for (int i = t; i < BB * 2 * HH; i += 256) sf[i] = feat[c * BB * 2 * HH + i];
    __syncthreads();

    const int b  = t >> 3;          // 8 threads per batch row
    const int ob = (t & 7) * 8;     // each thread: 8 output classes
    float acc[8];
#pragma unroll
    for (int o = 0; o < 8; ++o) acc[o] = bfc[ob + o];
    const float* fb = &sf[b * 2 * HH];
    for (int k = 0; k < 2 * HH; ++k) {
        float f = fb[k];
#pragma unroll
        for (int o = 0; o < 8; ++o)
            acc[o] = fmaf(f, Wfc[(ob + o) * 2 * HH + k], acc[o]);
    }
#pragma unroll
    for (int o = 0; o < 8; ++o) logits[b][ob + o] = fmaxf(acc[o], 0.0f);
    __syncthreads();

    if (t < BB) {
        float m = -1e30f;
        for (int o = 0; o < OO; ++o) m = fmaxf(m, logits[t][o]);
        float ssum = 0.0f;
        for (int o = 0; o < OO; ++o) ssum += __expf(logits[t][o] - m);
        float inv = 1.0f / ssum;
        float* op = out + (t * CC + c) * OO;   // output is (B, C, O)
        for (int o = 0; o < OO; ++o) op[o] = __expf(logits[t][o] - m) * inv;
    }
}

extern "C" void kernel_launch(void* const* d_in, const int* in_sizes, int n_in,
                              void* d_out, int out_size, void* d_ws, size_t ws_size,
                              hipStream_t stream) {
    (void)in_sizes; (void)n_in; (void)out_size; (void)ws_size;
    const float* x     = (const float*)d_in[0];
    const float* hprev = (const float*)d_in[1];
    const float* Wih_f = (const float*)d_in[2];
    const float* Whh_f = (const float*)d_in[3];
    const float* bih_f = (const float*)d_in[4];
    const float* bhh_f = (const float*)d_in[5];
    const float* Wih_b = (const float*)d_in[6];
    const float* Whh_b = (const float*)d_in[7];
    const float* bih_b = (const float*)d_in[8];
    const float* bhh_b = (const float*)d_in[9];
    const float* Wfc   = (const float*)d_in[10];
    const float* bfc   = (const float*)d_in[11];

    float* feat = (float*)d_ws;              // C*B*2H floats = 4 MB scratch
    float* out  = (float*)d_out;             // (B, C, O) float32

    // Two independent recurrent chains -> two persistent workgroups.
    gru_chain_kernel<<<2, 256, 0, stream>>>(x, hprev,
                                            Wih_f, Whh_f, bih_f, bhh_f,
                                            Wih_b, Whh_b, bih_b, bhh_b, feat);
    fc_softmax_kernel<<<CC, 256, 0, stream>>>(feat, Wfc, bfc, out);
}
